// ToProxy2_76312978915730
// MI455X (gfx1250) — compile-verified, matrix operand order checked by reading, not run on hardware
//
#include <hip/hip_runtime.h>
#include <hip/hip_bf16.h>

typedef unsigned short u16;
typedef __attribute__((ext_vector_type(8)))  unsigned short u16x8;
typedef __attribute__((ext_vector_type(16))) __bf16        v16bf;
typedef __attribute__((ext_vector_type(8)))  float         v8f;

#define C_      512
#define HEADS_  8
#define AGENT_  64
#define NTOK_   4096
#define NB_     8
#define ROWS_   (NB_ * NTOK_)   // 32768

__device__ __forceinline__ u16 f2bf(float f) {
    unsigned u = __float_as_uint(f);
    u = u + 0x7FFFu + ((u >> 16) & 1u);
    return (u16)(u >> 16);
}

struct FragBF { u16x8 lo, hi; };

__device__ __forceinline__ v16bf frag_cast(FragBF f) {
    union { FragBF a; v16bf b; } u; u.a = f; return u.b;
}

// Load a 16(row) x 32(k) bf16 WMMA operand fragment from a row-major buffer.
// Matches the CDNA5 16-bit A/B layout: lane<16 -> K in [0..7]+[16..23],
// lane>=16 -> K in [8..15]+[24..31], row = lane&15.
__device__ __forceinline__ v16bf load_frag(const u16* base, int row0, int stride,
                                           int k0, int lane) {
    int r = lane & 15, ko = (lane >> 4) * 8;
    const u16* p = base + (size_t)(row0 + r) * stride + k0 + ko;
    FragBF f;
    f.lo = *(const u16x8*)p;
    f.hi = *(const u16x8*)(p + 16);
    return frag_cast(f);
}

// Transposed fragment: operand rows index the COLUMN of a row-major [k][stride]
// buffer (used to feed V^T to the P@V WMMA from an LDS-staged V tile).
__device__ __forceinline__ v16bf load_frag_T(const u16* base, int d0, int stride, int lane) {
    int d = d0 + (lane & 15), ko = (lane >> 4) * 8;
    union { u16 s[16]; v16bf v; } u;
#pragma unroll
    for (int i = 0; i < 8; ++i) u.s[i]     = base[(ko + i) * stride + d];
#pragma unroll
    for (int i = 0; i < 8; ++i) u.s[8 + i] = base[(16 + ko + i) * stride + d];
    return u.v;
}

__device__ __forceinline__ v8f wmma_bf16(v16bf a, v16bf b, v8f c) {
    return __builtin_amdgcn_wmma_f32_16x16x32_bf16(false, a, false, b, (short)0, c,
                                                   false, false);
}

__device__ __forceinline__ v8f zero8() {
    v8f z;
#pragma unroll
    for (int i = 0; i < 8; ++i) z[i] = 0.f;
    return z;
}

// Per-wave GEMM: D(32x64) = A(32x512, LDS) * B^T(64x512 slice of wt), K=512.
__device__ __forceinline__ void gemm_32x64(const u16* sIn, const u16* wt, int colbase,
                                           int lane, v8f (&acc)[2][4]) {
#pragma unroll
    for (int rt = 0; rt < 2; ++rt)
#pragma unroll
        for (int ct = 0; ct < 4; ++ct) acc[rt][ct] = zero8();
#pragma unroll 2
    for (int kk = 0; kk < C_; kk += 32) {
        v16bf a0 = load_frag(sIn, 0,  C_, kk, lane);
        v16bf a1 = load_frag(sIn, 16, C_, kk, lane);
#pragma unroll
        for (int ct = 0; ct < 4; ++ct) {
            v16bf b = load_frag(wt, colbase + ct * 16, C_, kk, lane);
            acc[0][ct] = wmma_bf16(a0, b, acc[0][ct]);
            acc[1][ct] = wmma_bf16(a1, b, acc[1][ct]);
        }
    }
}

// ---------------------------------------------------------------- prep kernels

__global__ void k_transpose(const float* __restrict__ in, u16* __restrict__ out,
                            int K, int N) {
    int idx = blockIdx.x * 256 + threadIdx.x;
    if (idx >= K * N) return;
    int k = idx / N, n = idx % N;
    out[n * K + k] = f2bf(in[idx]);
}

__global__ void k_zero(float* __restrict__ p, int n) {
    int idx = blockIdx.x * 256 + threadIdx.x;
    if (idx < n) p[idx] = 0.f;
}

// k1[a][c] = lkv@w_kvy(:, :512)+b ; v1t[c][a] = lkv@w_kvy(:, 512:)+b ; q2g = lq@w_q+b
__global__ void k_prep(const float* __restrict__ lq, const float* __restrict__ lkv,
                       const float* __restrict__ w_q, const float* __restrict__ b_q,
                       const float* __restrict__ w_kvy, const float* __restrict__ b_kvy,
                       u16* __restrict__ k1t, u16* __restrict__ v1t, u16* __restrict__ q2g) {
    int idx = blockIdx.x * 256 + threadIdx.x;
    if (idx >= 3 * AGENT_ * C_) return;
    int which = idx / (AGENT_ * C_);
    int ii = idx & (AGENT_ * C_ - 1);
    int a = ii >> 9, c = ii & 511;
    float acc;
    if (which == 0) {
        acc = b_kvy[c];
        for (int k = 0; k < C_; ++k) acc += lkv[a * C_ + k] * w_kvy[k * 1024 + c];
        k1t[a * C_ + c] = f2bf(acc);
    } else if (which == 1) {
        acc = b_kvy[512 + c];
        for (int k = 0; k < C_; ++k) acc += lkv[a * C_ + k] * w_kvy[k * 1024 + 512 + c];
        v1t[c * AGENT_ + a] = f2bf(acc);
    } else {
        acc = b_q[c];
        for (int k = 0; k < C_; ++k) acc += lq[a * C_ + k] * w_q[k * C_ + c];
        q2g[a * C_ + c] = f2bf(acc);
    }
}

// ---------------------------------------------------------------- stage 1

__global__ __launch_bounds__(256) void k_stage1(
    const float* __restrict__ x,
    const u16* __restrict__ wq2t, const float* __restrict__ bq2,
    const u16* __restrict__ k1t, const u16* __restrict__ v1t,
    const float* __restrict__ pe1,
    const u16* __restrict__ f1w1t, const float* __restrict__ f1b1,
    const u16* __restrict__ f1w2t, const float* __restrict__ f1b2,
    const u16* __restrict__ wkvxt, const float* __restrict__ bkvx,
    float* __restrict__ mskip, u16* __restrict__ k2, u16* __restrict__ v2,
    float* __restrict__ vsum) {
    __shared__ __align__(16) u16 sA[32 * C_];
    __shared__ __align__(16) u16 sB[32 * C_];
    const int tid  = threadIdx.x;
    const int lane = tid & 31, w = tid >> 5;
    const int lo = lane & 15, hi4 = lane >> 4;
    const int r0 = blockIdx.x * 32;
    const int batch = r0 >> 12;
    const int cb = w * 64;

    // x tile -> bf16 LDS
#pragma unroll 4
    for (int i = 0; i < 64; ++i) {
        int e = i * 256 + tid;
        sA[e] = f2bf(x[(size_t)r0 * C_ + e]);
    }
    __syncthreads();

    v8f acc[2][4];

    // GEMM1: q = x @ w_q2 + b  -> sB
    gemm_32x64(sA, wq2t, cb, lane, acc);
#pragma unroll
    for (int rt = 0; rt < 2; ++rt)
#pragma unroll
        for (int ct = 0; ct < 4; ++ct) {
            int n = cb + ct * 16 + lo;
            float bv = bq2[n];
#pragma unroll
            for (int j = 0; j < 8; ++j) {
                int m = j + 8 * hi4 + 16 * rt;
                sB[m * C_ + n] = f2bf(acc[rt][ct][j] + bv);
            }
        }
    __syncthreads();  // all waves done reading sA; safe to overwrite

    // ---- per-head attention (head == wave id) ----
    {
        const float scale = 0.125f;
        v8f s[2][4];
#pragma unroll
        for (int rt = 0; rt < 2; ++rt)
#pragma unroll
            for (int ct = 0; ct < 4; ++ct) s[rt][ct] = zero8();
#pragma unroll
        for (int kk = 0; kk < 64; kk += 32) {
            v16bf a0 = load_frag(sB, 0,  C_, cb + kk, lane);
            v16bf a1 = load_frag(sB, 16, C_, cb + kk, lane);
#pragma unroll
            for (int ct = 0; ct < 4; ++ct) {
                v16bf b = load_frag(k1t, ct * 16, C_, cb + kk, lane);
                s[0][ct] = wmma_bf16(a0, b, s[0][ct]);
                s[1][ct] = wmma_bf16(a1, b, s[1][ct]);
            }
        }
        float pe1v[4];
#pragma unroll
        for (int ct = 0; ct < 4; ++ct) pe1v[ct] = pe1[w * 64 + ct * 16 + lo];

        // softmax over the 64 agent columns + pe1
#pragma unroll
        for (int rt = 0; rt < 2; ++rt)
#pragma unroll
            for (int j = 0; j < 8; ++j) {
                float v0 = s[rt][0][j] * scale, v1 = s[rt][1][j] * scale;
                float v2l = s[rt][2][j] * scale, v3 = s[rt][3][j] * scale;
                float mx = fmaxf(fmaxf(v0, v1), fmaxf(v2l, v3));
                mx = fmaxf(mx, __shfl_xor(mx, 1));
                mx = fmaxf(mx, __shfl_xor(mx, 2));
                mx = fmaxf(mx, __shfl_xor(mx, 4));
                mx = fmaxf(mx, __shfl_xor(mx, 8));
                v0 = __expf(v0 - mx); v1 = __expf(v1 - mx);
                v2l = __expf(v2l - mx); v3 = __expf(v3 - mx);
                float sm = v0 + v1 + v2l + v3;
                sm += __shfl_xor(sm, 1);
                sm += __shfl_xor(sm, 2);
                sm += __shfl_xor(sm, 4);
                sm += __shfl_xor(sm, 8);
                float inv = 1.f / sm;
                s[rt][0][j] = v0 * inv + pe1v[0];
                s[rt][1][j] = v1 * inv + pe1v[1];
                s[rt][2][j] = v2l * inv + pe1v[2];
                s[rt][3][j] = v3 * inv + pe1v[3];
            }
        // store P (bf16) into wave-private slice of sA
#pragma unroll
        for (int rt = 0; rt < 2; ++rt)
#pragma unroll
            for (int ct = 0; ct < 4; ++ct)
#pragma unroll
                for (int j = 0; j < 8; ++j) {
                    int m = j + 8 * hi4 + 16 * rt;
                    sA[m * C_ + cb + ct * 16 + lo] = f2bf(s[rt][ct][j]);
                }
        // O = P @ V1
        v8f o[2][4];
#pragma unroll
        for (int rt = 0; rt < 2; ++rt)
#pragma unroll
            for (int ct = 0; ct < 4; ++ct) o[rt][ct] = zero8();
#pragma unroll
        for (int kk = 0; kk < 64; kk += 32) {
            v16bf a0 = load_frag(sA, 0,  C_, cb + kk, lane);
            v16bf a1 = load_frag(sA, 16, C_, cb + kk, lane);
#pragma unroll
            for (int ct = 0; ct < 4; ++ct) {
                v16bf b = load_frag(v1t, cb + ct * 16, AGENT_, kk, lane);
                o[0][ct] = wmma_bf16(a0, b, o[0][ct]);
                o[1][ct] = wmma_bf16(a1, b, o[1][ct]);
            }
        }
#pragma unroll
        for (int rt = 0; rt < 2; ++rt)
#pragma unroll
            for (int ct = 0; ct < 4; ++ct)
#pragma unroll
                for (int j = 0; j < 8; ++j) {
                    int m = j + 8 * hi4 + 16 * rt;
                    sA[m * C_ + cb + ct * 16 + lo] = f2bf(o[rt][ct][j]);
                }
    }
    __syncthreads();

    // GEMM2: h1 = silu(O @ f1w1 + b1) -> sB
    gemm_32x64(sA, f1w1t, cb, lane, acc);
#pragma unroll
    for (int rt = 0; rt < 2; ++rt)
#pragma unroll
        for (int ct = 0; ct < 4; ++ct) {
            int n = cb + ct * 16 + lo;
            float bv = f1b1[n];
#pragma unroll
            for (int j = 0; j < 8; ++j) {
                int m = j + 8 * hi4 + 16 * rt;
                float v = acc[rt][ct][j] + bv;
                v = v / (1.f + __expf(-v));
                sB[m * C_ + n] = f2bf(v);
            }
        }
    __syncthreads();

    // GEMM3: mskip = h1 @ f1w2 + b2 + x  -> global f32 + sA bf16
    gemm_32x64(sB, f1w2t, cb, lane, acc);
#pragma unroll
    for (int rt = 0; rt < 2; ++rt)
#pragma unroll
        for (int ct = 0; ct < 4; ++ct) {
            int n = cb + ct * 16 + lo;
            float bv = f1b2[n];
#pragma unroll
            for (int j = 0; j < 8; ++j) {
                int m = j + 8 * hi4 + 16 * rt;
                float v = acc[rt][ct][j] + bv + x[(size_t)(r0 + m) * C_ + n];
                mskip[(size_t)(r0 + m) * C_ + n] = v;
                sA[m * C_ + n] = f2bf(v);
            }
        }
    __syncthreads();

    // GEMM4: kv2 = mskip @ w_kvx + b (N=1024 in two passes)
#pragma unroll 1
    for (int p = 0; p < 2; ++p) {
        gemm_32x64(sA, wkvxt, p * 512 + cb, lane, acc);
        u16* dst = p ? v2 : k2;
        float csum[4] = {0.f, 0.f, 0.f, 0.f};
#pragma unroll
        for (int ct = 0; ct < 4; ++ct) {
            int n = cb + ct * 16 + lo;
            float bv = bkvx[p * 512 + n];
#pragma unroll
            for (int rt = 0; rt < 2; ++rt)
#pragma unroll
                for (int j = 0; j < 8; ++j) {
                    int m = j + 8 * hi4 + 16 * rt;
                    float v = acc[rt][ct][j] + bv;
                    dst[(size_t)(r0 + m) * C_ + n] = f2bf(v);
                    if (p) csum[ct] += v;
                }
        }
        if (p) {
#pragma unroll
            for (int ct = 0; ct < 4; ++ct) {
                float c = csum[ct] + __shfl_xor(csum[ct], 16);
                if (lane < 16) atomicAdd(&vsum[batch * C_ + cb + ct * 16 + lo], c);
            }
        }
    }
}

// ---------------------------------------------------------------- stage 2

__global__ __launch_bounds__(128) void k_stage2(
    const u16* __restrict__ q2g, const u16* __restrict__ k2,
    const u16* __restrict__ v2, const float* __restrict__ pe2,
    const float* __restrict__ vsum, float* __restrict__ fin) {
    __shared__ __align__(16) u16 sK[32 * 64];
    __shared__ __align__(16) u16 sV[32 * 64];
    __shared__ __align__(16) u16 sP[4][16 * 32];
    const int tid = threadIdx.x, lane = tid & 31, w = tid >> 5;
    const int lo = lane & 15, hi4 = lane >> 4;
    const int b = blockIdx.x >> 3, h = blockIdx.x & 7;
    const u16* k2b = k2 + (size_t)b * NTOK_ * C_;
    const u16* v2b = v2 + (size_t)b * NTOK_ * C_;
    const float scale = 0.125f;

    v16bf aq0 = load_frag(q2g, 16 * w, C_, h * 64 + 0,  lane);
    v16bf aq1 = load_frag(q2g, 16 * w, C_, h * 64 + 32, lane);

    v8f o[4];
#pragma unroll
    for (int c4 = 0; c4 < 4; ++c4) o[c4] = zero8();
    float m_run[8], l_run[8];
#pragma unroll
    for (int j = 0; j < 8; ++j) { m_run[j] = -1e30f; l_run[j] = 0.f; }

#pragma unroll 1
    for (int it = 0; it < NTOK_ / 32; ++it) {
        int tok0 = it * 32;
#pragma unroll
        for (int cidx = 0; cidx < 2; ++cidx) {
            int cch = tid + cidx * 128;             // 0..255 chunks of 8
            int row = cch >> 3, col8 = (cch & 7) * 8;
            *(u16x8*)&sK[row * 64 + col8] =
                *(const u16x8*)(k2b + (size_t)(tok0 + row) * C_ + h * 64 + col8);
            *(u16x8*)&sV[row * 64 + col8] =
                *(const u16x8*)(v2b + (size_t)(tok0 + row) * C_ + h * 64 + col8);
        }
        __syncthreads();

        v8f s[2];
        s[0] = zero8(); s[1] = zero8();
#pragma unroll
        for (int ct = 0; ct < 2; ++ct) {
            v16bf b0 = load_frag(sK, ct * 16, 64, 0,  lane);
            v16bf b1 = load_frag(sK, ct * 16, 64, 32, lane);
            s[ct] = wmma_bf16(aq0, b0, s[ct]);
            s[ct] = wmma_bf16(aq1, b1, s[ct]);
        }
        // online softmax over tokens
#pragma unroll
        for (int j = 0; j < 8; ++j) {
            float v0 = s[0][j] * scale, v1 = s[1][j] * scale;
            float t = fmaxf(v0, v1);
            t = fmaxf(t, __shfl_xor(t, 1));
            t = fmaxf(t, __shfl_xor(t, 2));
            t = fmaxf(t, __shfl_xor(t, 4));
            t = fmaxf(t, __shfl_xor(t, 8));
            float mnew = fmaxf(m_run[j], t);
            float corr = __expf(m_run[j] - mnew);
            m_run[j] = mnew;
            v0 = __expf(v0 - mnew); v1 = __expf(v1 - mnew);
            s[0][j] = v0; s[1][j] = v1;
            float ts = v0 + v1;
            ts += __shfl_xor(ts, 1);
            ts += __shfl_xor(ts, 2);
            ts += __shfl_xor(ts, 4);
            ts += __shfl_xor(ts, 8);
            l_run[j] = l_run[j] * corr + ts;
#pragma unroll
            for (int c4 = 0; c4 < 4; ++c4) o[c4][j] *= corr;
        }
        // P -> wave-private LDS, then O += P @ V
        u16* sPw = &sP[w][0];
#pragma unroll
        for (int ct = 0; ct < 2; ++ct)
#pragma unroll
            for (int j = 0; j < 8; ++j)
                sPw[(j + 8 * hi4) * 32 + ct * 16 + lo] = f2bf(s[ct][j]);
        v16bf ap = load_frag(sPw, 0, 32, 0, lane);
#pragma unroll
        for (int c4 = 0; c4 < 4; ++c4) {
            v16bf bv = load_frag_T(sV, c4 * 16, 64, lane);
            o[c4] = wmma_bf16(ap, bv, o[c4]);
        }
        __syncthreads();
    }

    // epilogue: normalize + pe2 * colsum(V)
#pragma unroll
    for (int c4 = 0; c4 < 4; ++c4) {
        int c = h * 64 + c4 * 16 + lo;
        float vs = vsum[b * C_ + c];
#pragma unroll
        for (int j = 0; j < 8; ++j) {
            int agent = 16 * w + j + 8 * hi4;
            float pv = pe2[h * 64 + agent];
            fin[((size_t)(b * 64 + agent)) * C_ + c] = o[c4][j] / l_run[j] + pv * vs;
        }
    }
}

// ---------------------------------------------------------------- stage-2 MLP

__global__ __launch_bounds__(256) void k_mlp2(
    const float* __restrict__ fin,
    const u16* __restrict__ f2w1t, const float* __restrict__ f2b1,
    const u16* __restrict__ f2w2t, const float* __restrict__ f2b2,
    float* __restrict__ fin2) {
    __shared__ __align__(16) u16 sA[32 * C_];
    __shared__ __align__(16) u16 sB[32 * C_];
    const int tid = threadIdx.x, lane = tid & 31, w = tid >> 5;
    const int lo = lane & 15, hi4 = lane >> 4;
    const int r0 = blockIdx.x * 32;
    const int cb = w * 64;

#pragma unroll 4
    for (int i = 0; i < 64; ++i) {
        int e = i * 256 + tid;
        sA[e] = f2bf(fin[(size_t)r0 * C_ + e]);
    }
    __syncthreads();

    v8f acc[2][4];
    gemm_32x64(sA, f2w1t, cb, lane, acc);
#pragma unroll
    for (int rt = 0; rt < 2; ++rt)
#pragma unroll
        for (int ct = 0; ct < 4; ++ct) {
            int n = cb + ct * 16 + lo;
            float bv = f2b1[n];
#pragma unroll
            for (int j = 0; j < 8; ++j) {
                int m = j + 8 * hi4 + 16 * rt;
                float v = acc[rt][ct][j] + bv;
                v = v / (1.f + __expf(-v));
                sB[m * C_ + n] = f2bf(v);
            }
        }
    __syncthreads();
    gemm_32x64(sB, f2w2t, cb, lane, acc);
#pragma unroll
    for (int rt = 0; rt < 2; ++rt)
#pragma unroll
        for (int ct = 0; ct < 4; ++ct) {
            int n = cb + ct * 16 + lo;
            float bv = f2b2[n];
#pragma unroll
            for (int j = 0; j < 8; ++j) {
                int m = j + 8 * hi4 + 16 * rt;
                fin2[(size_t)(r0 + m) * C_ + n] = acc[rt][ct][j] + bv;
            }
        }
}

// ---------------------------------------------------------------- final fuse

__global__ void k_final(const float* __restrict__ x, const float* __restrict__ mskip,
                        const float* __restrict__ fin2, float* __restrict__ out) {
    int idx = blockIdx.x * 256 + threadIdx.x;       // float4 index
    if (idx >= ROWS_ * C_ / 4) return;
    size_t e = (size_t)idx * 4;
    int c = (int)(e & 511);
    int row = (int)(e >> 9);
    int b = row >> 12, ij = row & 4095;
    int i = ij >> 6, j = ij & 63;
    int fr = (b << 6) + ((i >> 3) << 3) + (j >> 3);
    float4 a  = *(const float4*)&x[e];
    float4 mm = *(const float4*)&mskip[e];
    float4 f  = *(const float4*)&fin2[(size_t)fr * C_ + c];
    float4 r;
    r.x = a.x + mm.x + f.x;
    r.y = a.y + mm.y + f.y;
    r.z = a.z + mm.z + f.z;
    r.w = a.w + mm.w + f.w;
    *(float4*)&out[e] = r;
}

// ---------------------------------------------------------------- launch

extern "C" void kernel_launch(void* const* d_in, const int* in_sizes, int n_in,
                              void* d_out, int out_size, void* d_ws, size_t ws_size,
                              hipStream_t stream) {
    const float* x     = (const float*)d_in[0];
    const float* lq    = (const float*)d_in[1];
    const float* lkv   = (const float*)d_in[2];
    const float* pe1   = (const float*)d_in[3];
    const float* pe2   = (const float*)d_in[4];
    const float* w_q   = (const float*)d_in[5];
    const float* b_q   = (const float*)d_in[6];
    const float* w_kvx = (const float*)d_in[7];
    const float* b_kvx = (const float*)d_in[8];
    const float* w_kvy = (const float*)d_in[9];
    const float* b_kvy = (const float*)d_in[10];
    const float* w_q2  = (const float*)d_in[11];
    const float* b_q2  = (const float*)d_in[12];
    const float* f1w1  = (const float*)d_in[13];
    const float* f1b1  = (const float*)d_in[14];
    const float* f1w2  = (const float*)d_in[15];
    const float* f1b2  = (const float*)d_in[16];
    const float* f2w1  = (const float*)d_in[17];
    const float* f2b1  = (const float*)d_in[18];
    const float* f2w2  = (const float*)d_in[19];
    const float* f2b2  = (const float*)d_in[20];

    char* ws = (char*)d_ws;
    size_t off = 0;
    auto alloc = [&](size_t bytes) -> void* {
        void* p = ws + off;
        off += (bytes + 255) & ~(size_t)255;
        return p;
    };
    float* mskip = (float*)alloc((size_t)ROWS_ * C_ * 4);   // 64 MB
    u16*   k2    = (u16*)  alloc((size_t)ROWS_ * C_ * 2);   // 32 MB
    u16*   v2    = (u16*)  alloc((size_t)ROWS_ * C_ * 2);   // 32 MB
    float* fin   = (float*)alloc((size_t)NB_ * AGENT_ * C_ * 4);
    float* fin2  = (float*)alloc((size_t)NB_ * AGENT_ * C_ * 4);
    float* vsum  = (float*)alloc((size_t)NB_ * C_ * 4);
    u16* wq2t    = (u16*)alloc((size_t)C_ * C_ * 2);
    u16* f1w1t   = (u16*)alloc((size_t)C_ * C_ * 2);
    u16* f1w2t   = (u16*)alloc((size_t)C_ * C_ * 2);
    u16* wkvxt   = (u16*)alloc((size_t)1024 * C_ * 2);
    u16* f2w1t   = (u16*)alloc((size_t)C_ * C_ * 2);
    u16* f2w2t   = (u16*)alloc((size_t)C_ * C_ * 2);
    u16* k1t     = (u16*)alloc((size_t)AGENT_ * C_ * 2);
    u16* v1t     = (u16*)alloc((size_t)C_ * AGENT_ * 2);
    u16* q2g     = (u16*)alloc((size_t)AGENT_ * C_ * 2);

    // weight preprocessing
    k_transpose<<<(C_ * C_ + 255) / 256, 256, 0, stream>>>(w_q2, wq2t, C_, C_);
    k_transpose<<<(C_ * C_ + 255) / 256, 256, 0, stream>>>(f1w1, f1w1t, C_, C_);
    k_transpose<<<(C_ * C_ + 255) / 256, 256, 0, stream>>>(f1w2, f1w2t, C_, C_);
    k_transpose<<<(C_ * 1024 + 255) / 256, 256, 0, stream>>>(w_kvx, wkvxt, C_, 1024);
    k_transpose<<<(C_ * C_ + 255) / 256, 256, 0, stream>>>(f2w1, f2w1t, C_, C_);
    k_transpose<<<(C_ * C_ + 255) / 256, 256, 0, stream>>>(f2w2, f2w2t, C_, C_);
    k_zero<<<(NB_ * C_ + 255) / 256, 256, 0, stream>>>(vsum, NB_ * C_);
    k_prep<<<(3 * AGENT_ * C_ + 255) / 256, 256, 0, stream>>>(
        lq, lkv, w_q, b_q, w_kvy, b_kvy, k1t, v1t, q2g);

    // stage 1
    k_stage1<<<ROWS_ / 32, 256, 0, stream>>>(
        x, wq2t, b_q2, k1t, v1t, pe1, f1w1t, f1b1, f1w2t, f1b2,
        wkvxt, b_kvx, mskip, k2, v2, vsum);

    // stage 2
    k_stage2<<<NB_ * HEADS_, 128, 0, stream>>>(q2g, k2, v2, pe2, vsum, fin);

    // stage-2 MLP
    k_mlp2<<<NB_ * AGENT_ / 32, 256, 0, stream>>>(fin, f2w1t, f2b1, f2w2t, f2b2, fin2);

    // upsample + residuals
    k_final<<<(ROWS_ * C_ / 4 + 255) / 256, 256, 0, stream>>>(
        x, mskip, fin2, (float*)d_out);
}
